// DCNv3_8297876816697
// MI455X (gfx1250) — matrix operand order repaired and checked by reading, not compiled, not run on hardware
//
#include <hip/hip_runtime.h>
#include <cstdint>

typedef __bf16 bf16_t;
typedef __attribute__((ext_vector_type(16))) __bf16 v16bf;
typedef __attribute__((ext_vector_type(8)))  __bf16 v8bf;
typedef __attribute__((ext_vector_type(8)))  float  v8f;

// Types for the async global->LDS builtin: it takes (int4 AS1*, int4 AS3*, imm, imm)
// with GCC-style vector typing (per hipcc diagnostic).
typedef int v4i_t __attribute__((vector_size(16)));
typedef __attribute__((address_space(1))) v4i_t gv4i_t;   // global int4
typedef __attribute__((address_space(3))) v4i_t lv4i_t;   // LDS int4

#define L_TOT   16384      // N*H*W
#define C_DIM   256
#define G_DIM   8
#define CG_DIM  32
#define P_DIM   9

#define LDS_ROW 264        // 256 bf16 + 8 pad -> 528 B row (132 dw, conflict-free)

// ---------------------------------------------------------------------------
// fp32 -> bf16 elementwise convert
// ---------------------------------------------------------------------------
__global__ __launch_bounds__(256) void f2bf_kernel(const float* __restrict__ in,
                                                   bf16_t* __restrict__ out, int n) {
  int i = blockIdx.x * blockDim.x + threadIdx.x;
  if (i < n) out[i] = (bf16_t)in[i];
}

// ---------------------------------------------------------------------------
// Weight transpose + convert: W [256 x Nc] fp32 -> Wt [Nc x 256] bf16
// ---------------------------------------------------------------------------
__global__ __launch_bounds__(256) void wtrans_kernel(const float* __restrict__ W,
                                                     bf16_t* __restrict__ Wt, int Nc) {
  int i = blockIdx.x * blockDim.x + threadIdx.x;   // over Nc*256
  if (i >= Nc * 256) return;
  int n = i >> 8;
  int k = i & 255;
  Wt[n * 256 + k] = (bf16_t)W[k * Nc + n];
}

// ---------------------------------------------------------------------------
// Depthwise 3x3 conv (pad 1) + BN(inference) + SiLU, NHWC, output bf16
// ---------------------------------------------------------------------------
__global__ __launch_bounds__(256) void dwconv_bn_silu_kernel(
    const float* __restrict__ x, const float* __restrict__ dwk,
    const float* __restrict__ bn_g, const float* __restrict__ bn_b,
    const float* __restrict__ bn_m, const float* __restrict__ bn_v,
    bf16_t* __restrict__ x1bf) {
  int idx = blockIdx.x * blockDim.x + threadIdx.x;   // over L*C
  if (idx >= L_TOT * C_DIM) return;
  int c   = idx & 255;
  int pix = idx >> 8;
  int w = pix & 63;
  int h = (pix >> 6) & 63;
  int n = pix >> 12;
  float acc = 0.f;
#pragma unroll
  for (int ky = 0; ky < 3; ++ky) {
    int yy = h + ky - 1;
    if (yy < 0 || yy >= 64) continue;
#pragma unroll
    for (int kx = 0; kx < 3; ++kx) {
      int xx = w + kx - 1;
      if (xx < 0 || xx >= 64) continue;
      acc += x[(((n * 64 + yy) * 64 + xx) << 8) + c] * dwk[c * 9 + ky * 3 + kx];
    }
  }
  float inv = rsqrtf(bn_v[c] + 1e-5f);
  float y = bn_g[c] * (acc - bn_m[c]) * inv + bn_b[c];
  float s = y / (1.f + __expf(-y));
  x1bf[idx] = (bf16_t)s;
}

// ---------------------------------------------------------------------------
// WMMA GEMM (templated on NC): Out[M x NC] = A[M x 256] * Bt[NC x 256]^T + bias
// The weight panel Bt (shared by ALL waves) is staged once per workgroup into
// LDS via async global->LDS copies (ASYNCcnt), padded to a conflict-free row
// stride, then the 8 waves feed v_wmma_f32_16x16x32_bf16 from ds_load_b128.
// One wave per 16-row output panel; N-tile loop fully unrolled.
// ---------------------------------------------------------------------------
template <int NC>
__global__ __launch_bounds__(256) void gemm_wmma_bf16_kernel(
    const bf16_t* __restrict__ A, const bf16_t* __restrict__ Bt,
    const float* __restrict__ bias, float* __restrict__ Out, int M) {
  constexpr int KDIM = 256;
  constexpr int NT = (NC + 15) / 16;
  extern __shared__ __align__(16) char dynsmem[];
  bf16_t* smem = (bf16_t*)dynsmem;               // [NC][LDS_ROW] bf16

  // ---- cooperative stage: Bt [NC x 256] -> LDS [NC x LDS_ROW] -------------
  {
    const int tid = threadIdx.x;
    constexpr int CHUNKS = NC * 32;              // 16B chunks (32 per row)
    for (int i = tid; i < CHUNKS; i += 256) {
      int row = i >> 5;
      int cin = i & 31;                          // 16B chunk within row
      const char* gsrc = (const char*)Bt + (size_t)row * 512 + cin * 16;
      char*       ldst = (char*)smem + (size_t)row * (LDS_ROW * 2) + cin * 16;
#if __has_builtin(__builtin_amdgcn_global_load_async_to_lds_b128)
      __builtin_amdgcn_global_load_async_to_lds_b128(
          (gv4i_t*)(uintptr_t)gsrc,
          (lv4i_t*)(uint32_t)(uintptr_t)ldst,    // low 32 bits = LDS offset
          0, 0);
#else
      *(v8bf*)ldst = *(const v8bf*)gsrc;         // fallback: load + ds_store
#endif
    }
#if __has_builtin(__builtin_amdgcn_global_load_async_to_lds_b128)
#if __has_builtin(__builtin_amdgcn_s_wait_asynccnt)
    __builtin_amdgcn_s_wait_asynccnt(0);
#else
    asm volatile("s_wait_asynccnt 0" ::: "memory");
#endif
#endif
    __syncthreads();
  }

  int wave = (int)((blockIdx.x * blockDim.x + threadIdx.x) >> 5);
  int lane = threadIdx.x & 31;
  if (wave >= (M >> 4)) return;        // wave-uniform (grid exact): EXEC all-ones
  int m0   = wave << 4;
  int half = lane >> 4;                // 0 or 1
  int l16  = lane & 15;

  // ---- A panel: row m = m0+l16 ; step s covers K=[32s,32s+32) -------------
  // ISA 16-bit A 16x32 layout: lane-half selects K sub-blocks of 8.
  v16bf a[8];
  const bf16_t* arow = A + (size_t)(m0 + l16) * KDIM;
#pragma unroll
  for (int s = 0; s < 8; ++s) {
    int k0 = s * 32;
    v8bf alo = *(const v8bf*)(arow + k0 + half * 8);
    v8bf ahi = *(const v8bf*)(arow + k0 + 16 + half * 8);
    v16bf av;
#pragma unroll
    for (int e = 0; e < 8; ++e) { av[e] = alo[e]; av[e + 8] = ahi[e]; }
    a[s] = av;
  }

  // ---- N-tile sweep: B fragments from LDS ---------------------------------
#pragma unroll
  for (int nt = 0; nt < NT; ++nt) {
    int col = nt * 16 + l16;
    bool ok = (NC % 16 == 0) || (nt < NT - 1) || (col < NC);
    int bcol = col;
    if ((NC % 16) && nt == NT - 1 && bcol >= NC) bcol = NC - 1;  // clamp
    const bf16_t* brow = smem + (size_t)bcol * LDS_ROW + half * 8;
    float bv = ok ? bias[col] : 0.f;
    v8f acc = { bv, bv, bv, bv, bv, bv, bv, bv };
#pragma unroll
    for (int s = 0; s < 8; ++s) {
      v8bf blo = *(const v8bf*)(brow + s * 32);       // ds_load_b128
      v8bf bhi = *(const v8bf*)(brow + s * 32 + 16);  // ds_load_b128
      v16bf b;
#pragma unroll
      for (int e = 0; e < 8; ++e) { b[e] = blo[e]; b[e + 8] = bhi[e]; }
      acc = __builtin_amdgcn_wmma_f32_16x16x32_bf16(
          /*neg_a=*/false, a[s], /*neg_b=*/false, b,
          /*c_mod=*/(short)0, acc, /*reuse_a=*/false, /*reuse_b=*/false);
    }
    if (ok) {
      // C/D layout: lanes16-31 -> rows M+8; NC compile-time => imm offsets
      float* op = Out + (size_t)(m0 + half * 8) * NC + col;
#pragma unroll
      for (int r = 0; r < 8; ++r) op[(size_t)r * NC] = acc[r];
    }
  }
}

// ---------------------------------------------------------------------------
// Softmax over P=9 per (pixel, g), in place on [L, 72]
// ---------------------------------------------------------------------------
__global__ __launch_bounds__(256) void softmax9_kernel(float* __restrict__ m) {
  int t = blockIdx.x * blockDim.x + threadIdx.x;     // over L*G
  if (t >= L_TOT * G_DIM) return;
  float* p = m + (size_t)t * P_DIM;
  float v[P_DIM];
  float mx = -3.4e38f;
#pragma unroll
  for (int i = 0; i < P_DIM; ++i) { v[i] = p[i]; mx = fmaxf(mx, v[i]); }
  float s = 0.f;
#pragma unroll
  for (int i = 0; i < P_DIM; ++i) { v[i] = __expf(v[i] - mx); s += v[i]; }
  float inv = 1.f / s;
#pragma unroll
  for (int i = 0; i < P_DIM; ++i) p[i] = v[i] * inv;
}

// ---------------------------------------------------------------------------
// DCNv3 core: one wave per (pixel, group), lane = channel within group (Cg=32).
// ---------------------------------------------------------------------------
__global__ __launch_bounds__(256) void dcn_core_kernel(
    const float* __restrict__ xp,    // [L, 256]
    const float* __restrict__ off,   // [L, 144]  (g, p, xy)
    const float* __restrict__ msk,   // [L, 72]   (g, p)
    bf16_t* __restrict__ corebf) {   // [L, 256]
  int wave = (int)((blockIdx.x * blockDim.x + threadIdx.x) >> 5);
  int lane = threadIdx.x & 31;
  if (wave >= L_TOT * G_DIM) return;
  int g   = wave & 7;
  int pix = wave >> 3;
  int w = pix & 63;
  int h = (pix >> 6) & 63;
  int n = pix >> 12;
  const float Hi = 66.f, Wi = 66.f;
  float refx = (1.5f + (float)w) / Wi;
  float refy = (1.5f + (float)h) / Hi;
  const float* op   = off + (size_t)pix * 144 + g * 18;
  const float* mp   = msk + (size_t)pix * 72 + g * 9;
  const float* base = xp + ((size_t)n * 4096) * C_DIM + g * CG_DIM + lane;
  float acc = 0.f;
#pragma unroll
  for (int p = 0; p < P_DIM; ++p) {
    float gx = (float)(p / 3 - 1) / Wi;    // kx outer
    float gy = (float)(p % 3 - 1) / Hi;    // ky inner
    float locx = refx + gx + op[2 * p + 0] / Wi;
    float locy = refy + gy + op[2 * p + 1] / Hi;
    float sx = 2.f * locx - 1.f;
    float sy = 2.f * locy - 1.f;
    float fxs = ((sx + 1.f) * Wi - 1.f) * 0.5f;   // padded-image coords
    float fys = ((sy + 1.f) * Hi - 1.f) * 0.5f;
    float x0f = floorf(fxs), y0f = floorf(fys);
    float fx = fxs - x0f, fy = fys - y0f;
    int x0 = (int)x0f, y0 = (int)y0f;
    float m = mp[p];
    float s00 = 0.f, s10 = 0.f, s01 = 0.f, s11 = 0.f;
    // nonzero only inside the unpadded interior (pad ring is zero)
    if (y0 >= 1 && y0 <= 64 && x0 >= 1 && x0 <= 64)
      s00 = base[(size_t)((y0 - 1) * 64 + (x0 - 1)) * C_DIM];
    if (y0 >= 1 && y0 <= 64 && x0 + 1 >= 1 && x0 + 1 <= 64)
      s10 = base[(size_t)((y0 - 1) * 64 + x0) * C_DIM];
    if (y0 + 1 >= 1 && y0 + 1 <= 64 && x0 >= 1 && x0 <= 64)
      s01 = base[(size_t)(y0 * 64 + (x0 - 1)) * C_DIM];
    if (y0 + 1 >= 1 && y0 + 1 <= 64 && x0 + 1 >= 1 && x0 + 1 <= 64)
      s11 = base[(size_t)(y0 * 64 + x0) * C_DIM];
    float v = s00 * (1.f - fx) * (1.f - fy) + s10 * fx * (1.f - fy) +
              s01 * (1.f - fx) * fy        + s11 * fx * fy;
    acc += m * v;
  }
  corebf[(size_t)pix * C_DIM + g * CG_DIM + lane] = (bf16_t)acc;
}

// ---------------------------------------------------------------------------
// Launcher
// ---------------------------------------------------------------------------
extern "C" void kernel_launch(void* const* d_in, const int* in_sizes, int n_in,
                              void* d_out, int out_size, void* d_ws, size_t ws_size,
                              hipStream_t stream) {
  (void)in_sizes; (void)n_in; (void)out_size; (void)ws_size;
  const float* x      = (const float*)d_in[0];
  const float* in_w   = (const float*)d_in[1];
  const float* in_b   = (const float*)d_in[2];
  const float* out_w  = (const float*)d_in[3];
  const float* out_b  = (const float*)d_in[4];
  const float* off_w  = (const float*)d_in[5];
  const float* off_b  = (const float*)d_in[6];
  const float* mask_w = (const float*)d_in[7];
  const float* mask_b = (const float*)d_in[8];
  const float* dw_k   = (const float*)d_in[9];
  const float* bn_g   = (const float*)d_in[10];
  const float* bn_b   = (const float*)d_in[11];
  const float* bn_m   = (const float*)d_in[12];
  const float* bn_v   = (const float*)d_in[13];
  float* out = (float*)d_out;

  char* ws = (char*)d_ws;
  bf16_t* x_bf    = (bf16_t*)(ws + 0);          //  8 MB  [L,256] bf16
  bf16_t* x1_bf   = (bf16_t*)(ws + 8388608);    //  8 MB  [L,256] bf16
  float*  xp      = (float*) (ws + 16777216);   // 16 MB  [L,256] f32
  float*  offb    = (float*) (ws + 33554432);   //  9 MB  [L,144] f32
  float*  maskb   = (float*) (ws + 42991616);   //  4.5MB [L,72]  f32
  bf16_t* core_bf = (bf16_t*)(ws + 47710208);   //  8 MB  [L,256] bf16
  bf16_t* in_wT   = (bf16_t*)(ws + 56098816);   // [256,256] bf16
  bf16_t* off_wT  = (bf16_t*)(ws + 56229888);   // [144,256] bf16
  bf16_t* mask_wT = (bf16_t*)(ws + 56303616);   // [72,256]  bf16
  bf16_t* out_wT  = (bf16_t*)(ws + 56340480);   // [256,256] bf16

  // 1) precision conversions
  f2bf_kernel<<<16384, 256, 0, stream>>>(x, x_bf, L_TOT * C_DIM);
  wtrans_kernel<<<256, 256, 0, stream>>>(in_w,   in_wT,   256);
  wtrans_kernel<<<144, 256, 0, stream>>>(off_w,  off_wT,  144);
  wtrans_kernel<<<72,  256, 0, stream>>>(mask_w, mask_wT, 72);
  wtrans_kernel<<<256, 256, 0, stream>>>(out_w,  out_wT,  256);

  // 2) depthwise conv + BN + SiLU -> x1 (bf16)
  dwconv_bn_silu_kernel<<<16384, 256, 0, stream>>>(x, dw_k, bn_g, bn_b, bn_m,
                                                   bn_v, x1_bf);

  // 3) WMMA GEMMs with LDS-staged weight panels
  //    1024 M-tiles -> 1024 waves -> 128 blocks of 256 threads (8 waves)
  gemm_wmma_bf16_kernel<256><<<128, 256, 256 * LDS_ROW * 2, stream>>>(
      x_bf,  in_wT,   in_b,   xp,    L_TOT);
  gemm_wmma_bf16_kernel<144><<<128, 256, 144 * LDS_ROW * 2, stream>>>(
      x1_bf, off_wT,  off_b,  offb,  L_TOT);
  gemm_wmma_bf16_kernel<72><<<128, 256, 72 * LDS_ROW * 2, stream>>>(
      x1_bf, mask_wT, mask_b, maskb, L_TOT);

  // 4) softmax over P per (pixel, g)
  softmax9_kernel<<<512, 256, 0, stream>>>(maskb);

  // 5) deformable bilinear gather core -> bf16
  dcn_core_kernel<<<16384, 256, 0, stream>>>(xp, offb, maskb, core_bf);

  // 6) output projection -> d_out (fp32)
  gemm_wmma_bf16_kernel<256><<<128, 256, 256 * LDS_ROW * 2, stream>>>(
      core_bf, out_wT, out_b, out, L_TOT);
}
